// dMaSIFConv_seg_23278722744328
// MI455X (gfx1250) — compile-verified, matrix operand order checked by reading, not run on hardware
//
#include <hip/hip_runtime.h>
#include <hip/hip_bf16.h>

#define NPTS   3072
#define OC     64
#define CUTS   8
#define NPLANE 9            // 8 G-channels + 1 window channel
#define JCHUNK 32
#define CWAVES 4
#define JSPLIT 4            // K-dimension split across blockIdx.y

typedef __attribute__((ext_vector_type(16))) __bf16        v16bf;
typedef __attribute__((ext_vector_type(8)))  float         v8f;
typedef __attribute__((ext_vector_type(8)))  unsigned short v8u;

union BF {
    v16bf v;
    struct { v8u lo, hi; } h;
};

// accurate RNE conversion (used in the cheap precompute kernel)
__device__ __forceinline__ unsigned short f2bf_rne(float x) {
    unsigned int b = __float_as_uint(x);
    unsigned int r = b + 0x7FFFu + ((b >> 16) & 1u);
    return (unsigned short)(r >> 16);
}
// fast bias-rounded conversion (hot path: 2 VALU ops)
__device__ __forceinline__ unsigned short f2bf_fast(float x) {
    return (unsigned short)((__float_as_uint(x) + 0x8000u) >> 16);
}

__device__ __forceinline__ float actf(float x, int mode) {
    if (mode == 1) return fmaxf(x, 0.0f);          // relu
    if (mode == 2) return x > 0.0f ? x : 0.2f * x; // leaky relu 0.2
    return x;
}

// ---------------------------------------------------------------------------
// zero buffers (stats, conv accumulator)
// ---------------------------------------------------------------------------
__global__ void zero_kernel(float* __restrict__ p, int n) {
    int i = blockIdx.x * blockDim.x + threadIdx.x;
    if (i < n) p[i] = 0.0f;
}

// ---------------------------------------------------------------------------
// Two stacked 64-wide linear layers with activations; optional group stats.
// ---------------------------------------------------------------------------
__global__ __launch_bounds__(256) void mlp2_kernel(
    const float* __restrict__ xin, int ind,
    const float* __restrict__ w1, const float* __restrict__ b1, int act1,
    const float* __restrict__ w2, const float* __restrict__ b2, int act2,
    float* __restrict__ out, float* __restrict__ stats)
{
    __shared__ float sf1[64 * 64];
    __shared__ float sred[8];
    const int tid   = threadIdx.x;
    const int rbase = blockIdx.x * 64;

    for (int e = tid; e < 64 * 64; e += 256) {
        int r = e >> 6, c = e & 63;
        const float* xr = xin + (size_t)(rbase + r) * ind;
        const float* wr = w1 + (size_t)c * ind;
        float a = b1[c];
        for (int d = 0; d < ind; ++d) a += xr[d] * wr[d];
        sf1[e] = actf(a, act1);
    }
    __syncthreads();

    float ls[4] = {0.f, 0.f, 0.f, 0.f};
    float lq[4] = {0.f, 0.f, 0.f, 0.f};
    for (int e = tid; e < 64 * 64; e += 256) {
        int r = e >> 6, c = e & 63;
        const float* fr = sf1 + r * 64;
        const float* wr = w2 + c * 64;
        float a = b2[c];
#pragma unroll 8
        for (int d = 0; d < 64; ++d) a += fr[d] * wr[d];
        a = actf(a, act2);
        out[(size_t)(rbase + r) * 64 + c] = a;
        if (stats) { int g = c >> 4; ls[g] += a; lq[g] += a * a; }
    }
    if (stats) {
        if (tid < 8) sred[tid] = 0.0f;
        __syncthreads();
#pragma unroll
        for (int g = 0; g < 4; ++g) {
            atomicAdd(&sred[g],     ls[g]);   // ds_add_f32
            atomicAdd(&sred[4 + g], lq[g]);
        }
        __syncthreads();
        if (tid < 8) atomicAdd(&stats[tid], sred[tid]); // global_atomic_add_f32
    }
}

// ---------------------------------------------------------------------------
// Apply GroupNorm to f2 and emit 9 bf16 B-planes, transposed: BcT[c][h][j]
// ---------------------------------------------------------------------------
__global__ __launch_bounds__(256) void bct_kernel(
    const float* __restrict__ f2, const float* __restrict__ stats,
    const float* __restrict__ gw, const float* __restrict__ gb,
    const float* __restrict__ cvw2, const float* __restrict__ cvb2,
    unsigned short* __restrict__ BcT)
{
    int idx = blockIdx.x * 256 + threadIdx.x;
    if (idx >= NPTS * OC) return;
    int j = idx >> 6, h = idx & 63, g = h >> 4;
    const float cnt = (float)(NPTS * 16);
    float mu  = stats[g] / cnt;
    float var = stats[4 + g] / cnt - mu * mu;
    float f = (f2[idx] - mu) * rsqrtf(var + 1e-5f) * gw[h] + gb[h];
#pragma unroll
    for (int c = 0; c < CUTS; ++c)
        BcT[(size_t)(c * OC + h) * NPTS + j] = f2bf_rne(f * cvw2[h * CUTS + c]);
    BcT[(size_t)(CUTS * OC + h) * NPTS + j] = f2bf_rne(f * cvb2[h]);
}

// ---------------------------------------------------------------------------
// GroupNorm only
// ---------------------------------------------------------------------------
__global__ __launch_bounds__(256) void gn_kernel(
    const float* __restrict__ f2, const float* __restrict__ stats,
    const float* __restrict__ gw, const float* __restrict__ gb,
    float* __restrict__ out)
{
    int idx = blockIdx.x * 256 + threadIdx.x;
    if (idx >= NPTS * OC) return;
    int h = idx & 63, g = h >> 4;
    const float cnt = (float)(NPTS * 16);
    float mu  = stats[g] / cnt;
    float var = stats[4 + g] / cnt - mu * mu;
    out[idx] = (f2[idx] - mu) * rsqrtf(var + 1e-5f) * gw[h] + gb[h];
}

// ---------------------------------------------------------------------------
// Skip path: xnew = xold @ lt_w.T + lt_b + t2
// ---------------------------------------------------------------------------
__global__ __launch_bounds__(256) void linadd_kernel(
    const float* __restrict__ xold, int ind,
    const float* __restrict__ ltw, const float* __restrict__ ltb,
    const float* __restrict__ t2, float* __restrict__ xnew)
{
    int idx = blockIdx.x * 256 + threadIdx.x;
    if (idx >= NPTS * OC) return;
    int r = idx >> 6, c = idx & 63;
    float a = ltb[c] + t2[idx];
    const float* xr = xold + (size_t)r * ind;
    const float* wr = ltw + (size_t)c * ind;
    for (int d = 0; d < ind; ++d) a += xr[d] * wr[d];
    xnew[idx] = a;
}

// ---------------------------------------------------------------------------
// O(N^2) point-conv as an implicit GEMM: M=3072 (i), N=64 (h), fused K=9*3072
// over (j,c). bf16 WMMA 16x16x32, f32 accumulate.
// grid = (N/16 i-tiles) x (JSPLIT k-slices); block = 4 waves, each streaming
// every 4th 32-wide j-chunk of its k-slice. Partials accumulated with
// global_atomic_add_f32 into convout.
// ---------------------------------------------------------------------------
__global__ __launch_bounds__(128, 1) void conv_kernel(
    const float* __restrict__ pts,      // N x 3
    const float* __restrict__ nuv,      // N x 3 x 3
    const float* __restrict__ cv_w1,    // 8 x 3
    const float* __restrict__ cv_b1,    // 8
    const unsigned short* __restrict__ BcT, // 9 x 64 x N  (bf16 bits)
    float* __restrict__ out)            // N x 64 (atomic accumulate)
{
    // per-wave A tiles in bf16 A-fragment row-major layout
    __shared__ __align__(16) unsigned short sA[CWAVES][NPLANE][16][JCHUNK];
    __shared__ float sRed[CWAVES][16 * 64];
    __shared__ float sI[16][12];        // scaled x (3) + nuv (9) for the i-tile

    const int tid   = threadIdx.x;
    const int wave  = tid >> 5;
    const int lane  = tid & 31;
    const int itile = blockIdx.x;
    const int jlo   = blockIdx.y * (NPTS / JCHUNK / JSPLIT);   // 24 chunks/slice
    const float inv = 0.07856742013183861f;  // 1/(sqrt(2)*RADIUS)

    if (tid < 16) {
        int i = itile * 16 + tid;
        sI[tid][0] = pts[i * 3 + 0] * inv;
        sI[tid][1] = pts[i * 3 + 1] * inv;
        sI[tid][2] = pts[i * 3 + 2] * inv;
#pragma unroll
        for (int k = 0; k < 9; ++k) sI[tid][3 + k] = nuv[i * 9 + k];
    }

    float cw[CUTS][3], cb[CUTS];   // uniform -> SGPRs
#pragma unroll
    for (int c = 0; c < CUTS; ++c) {
        cw[c][0] = cv_w1[c * 3 + 0];
        cw[c][1] = cv_w1[c * 3 + 1];
        cw[c][2] = cv_w1[c * 3 + 2];
        cb[c]    = cv_b1[c];
    }

    v8f acc[4] = {};    // 16x64 f32 partial accumulator = 4 h-tiles
    __syncthreads();

    const int m  = lane & 15;
    const int ko = (lane >> 4) * 8;   // K sub-block per A/B fragment layout

    for (int jb = jlo + wave; jb < jlo + NPTS / JCHUNK / JSPLIT; jb += CWAVES) {
        // ----- build A tile: lane owns column j = jb*32+lane, loops 16 rows
        const int j = jb * JCHUNK + lane;
        float xj0 = pts[j * 3 + 0] * inv;
        float xj1 = pts[j * 3 + 1] * inv;
        float xj2 = pts[j * 3 + 2] * inv;
        float nj0 = nuv[j * 9 + 0], nj1 = nuv[j * 9 + 1], nj2 = nuv[j * 9 + 2];

        __builtin_prefetch(&BcT[(size_t)m * NPTS + jb * JCHUNK], 0, 1);

#pragma unroll 4
        for (int i = 0; i < 16; ++i) {
            float d0 = xj0 - sI[i][0], d1 = xj1 - sI[i][1], d2 = xj2 - sI[i][2];
            float r2 = d0 * d0 + d1 * d1 + d2 * d2;
            float dot = sI[i][3] * nj0 + sI[i][4] * nj1 + sI[i][5] * nj2;
            float t  = 2.0f - dot;
            float wexp = __expf(-r2 * t * t);
            float X0 = sI[i][3] * d0 + sI[i][4]  * d1 + sI[i][5]  * d2;
            float X1 = sI[i][6] * d0 + sI[i][7]  * d1 + sI[i][8]  * d2;
            float X2 = sI[i][9] * d0 + sI[i][10] * d1 + sI[i][11] * d2;
#pragma unroll
            for (int c = 0; c < CUTS; ++c) {
                float g = fmaxf(cw[c][0] * X0 + cw[c][1] * X1 + cw[c][2] * X2 + cb[c], 0.0f);
                sA[wave][c][i][lane] = f2bf_fast(wexp * g);
            }
            sA[wave][CUTS][i][lane] = f2bf_fast(wexp);
        }
        // wave-private LDS tile: only intra-wave store->load ordering needed
        asm volatile("s_wait_dscnt 0x0" ::: "memory");

        // ----- 9 c-planes x 4 h-tiles WMMAs, K = this 32-wide j chunk
#pragma unroll
        for (int c = 0; c < NPLANE; ++c) {
            BF a;
            a.h.lo = *(const v8u*)&sA[wave][c][m][ko];
            a.h.hi = *(const v8u*)&sA[wave][c][m][ko + 16];
#pragma unroll
            for (int ht = 0; ht < 4; ++ht) {
                const unsigned short* bp =
                    BcT + (size_t)(c * OC + ht * 16 + m) * NPTS + jb * JCHUNK + ko;
                BF b;
                b.h.lo = *(const v8u*)bp;
                b.h.hi = *(const v8u*)(bp + 16);
                acc[ht] = __builtin_amdgcn_wmma_f32_16x16x32_bf16(
                    false, a.v, false, b.v, (short)0, acc[ht], false, false);
            }
        }
    }

    // ----- cross-wave reduction of partial accumulators
#pragma unroll
    for (int t = 0; t < 4; ++t)
#pragma unroll
        for (int r = 0; r < 8; ++r)
            sRed[wave][t * 256 + r * 32 + lane] = acc[t][r];
    __syncthreads();

    {   // wave w finalizes h-tile w; accumulate k-slice partial into out
        const int ht = wave;
#pragma unroll
        for (int r = 0; r < 8; ++r) {
            int o = ht * 256 + r * 32 + lane;
            float v = sRed[0][o] + sRed[1][o] + sRed[2][o] + sRed[3][o];
            int mrow = r + ((lane >> 4) ? 8 : 0);   // C/D f32 tile layout
            int ncol = lane & 15;
            atomicAdd(&out[(size_t)(itile * 16 + mrow) * OC + ht * 16 + ncol], v);
        }
    }
}

// ---------------------------------------------------------------------------
extern "C" void kernel_launch(void* const* d_in, const int* in_sizes, int n_in,
                              void* d_out, int out_size, void* d_ws, size_t ws_size,
                              hipStream_t stream)
{
    const float* features = (const float*)d_in[0];  // N x 16
    const float* points   = (const float*)d_in[1];  // N x 3
    const float* nuv      = (const float*)d_in[2];  // N x 3 x 3

    char* ws = (char*)d_ws;
    float*          f2      = (float*)(ws + 0x000000);
    float*          o2      = (float*)(ws + 0x0C0000);
    float*          gbuf    = (float*)(ws + 0x180000);
    float*          t2      = (float*)(ws + 0x240000);
    float*          convout = (float*)(ws + 0x300000);
    float*          xA      = (float*)(ws + 0x3C0000);
    float*          stats   = (float*)(ws + 0x480000);
    unsigned short* BcT     = (unsigned short*)(ws + 0x480400); // 9*64*N bf16

    const int L = 2;
    for (int l = 0; l < L; ++l) {
        const float* P[22];
        for (int k = 0; k < 22; ++k) P[k] = (const float*)d_in[3 + l * 22 + k];
        const float *in_w1 = P[0],  *in_b1 = P[1],  *in_w2 = P[2],  *in_b2 = P[3];
        const float *gn_in_w = P[4], *gn_in_b = P[5];
        const float *cv_w1 = P[6],  *cv_b1 = P[7],  *cv_w2 = P[8],  *cv_b2 = P[9];
        const float *out_w1 = P[10], *out_b1 = P[11], *out_w2 = P[12], *out_b2 = P[13];
        const float *gn_out_w = P[14], *gn_out_b = P[15];
        const float *ll_w1 = P[16], *ll_b1 = P[17], *ll_w2 = P[18], *ll_b2 = P[19];
        const float *lt_w = P[20],  *lt_b = P[21];

        const float* xin = (l == 0) ? features : xA;
        const int    ind = (l == 0) ? 16 : 64;
        float*       xout = (l == L - 1) ? (float*)d_out : xA;

        zero_kernel<<<1, 32, 0, stream>>>(stats, 8);
        mlp2_kernel<<<NPTS / 64, 256, 0, stream>>>(xin, ind, in_w1, in_b1, 2,
                                                   in_w2, in_b2, 2, f2, stats);
        bct_kernel<<<NPTS * OC / 256, 256, 0, stream>>>(f2, stats, gn_in_w, gn_in_b,
                                                        cv_w2, cv_b2, BcT);
        zero_kernel<<<NPTS * OC / 256, 256, 0, stream>>>(convout, NPTS * OC);
        {
            dim3 grid(NPTS / 16, JSPLIT);
            conv_kernel<<<grid, 128, 0, stream>>>(points, nuv, cv_w1, cv_b1,
                                                  BcT, convout);
        }
        zero_kernel<<<1, 32, 0, stream>>>(stats, 8);
        mlp2_kernel<<<NPTS / 64, 256, 0, stream>>>(convout, 64, out_w1, out_b1, 2,
                                                   out_w2, out_b2, 2, o2, stats);
        gn_kernel<<<NPTS * OC / 256, 256, 0, stream>>>(o2, stats, gn_out_w, gn_out_b, gbuf);
        mlp2_kernel<<<NPTS / 64, 256, 0, stream>>>(gbuf, 64, ll_w1, ll_b1, 1,
                                                   ll_w2, ll_b2, 0, t2, nullptr);
        linadd_kernel<<<NPTS * OC / 256, 256, 0, stream>>>(xin, ind, lt_w, lt_b, t2, xout);
    }
}
